// GATLayer_3779571220517
// MI455X (gfx1250) — matrix operand order repaired and checked by reading, not compiled, not run on hardware
//
#include <hip/hip_runtime.h>
#include <hip/hip_bf16.h>

// ---------------------------------------------------------------------------
// GAT layer for MI455X (gfx1250, wave32). Query length == 1 lets us factor
// attention:  scores = val . (Wk_h^T q_h),  out = (sum_l a_l val_l) Wv_h + bv.
// Heavy work = 5 [N,128]x[128,128] GEMMs via v_wmma_f32_16x16x32_f16, with the
// weight (B^T) tile staged into LDS per-workgroup via the Tensor Data Mover.
// ---------------------------------------------------------------------------

typedef __attribute__((ext_vector_type(16))) _Float16 v16h;
typedef __attribute__((ext_vector_type(8)))  _Float16 v8h;
typedef __attribute__((ext_vector_type(4)))  _Float16 v4h;
typedef __attribute__((ext_vector_type(8)))  float    v8f;
typedef __attribute__((ext_vector_type(4)))  unsigned int u32x4;
typedef __attribute__((ext_vector_type(8)))  int      i32x8;
typedef __attribute__((ext_vector_type(4)))  int      i32x4;

#define D128 128

// ---- Tensor Data Mover availability / arity detection ---------------------
#if defined(__has_builtin)
#  if __has_builtin(__builtin_amdgcn_tensor_load_to_lds) && \
      __has_builtin(__builtin_amdgcn_s_wait_tensorcnt)
#    define USE_TDM 1
#  else
#    define USE_TDM 0
#  endif
#else
#  define USE_TDM 0
#endif
#if USE_TDM
#  if __has_include(<hip/amd_detail/amd_gfx1250_TDM.h>) || (__clang_major__ >= 23)
#    define TDM_6ARG 1
#  else
#    define TDM_6ARG 0
#  endif
#endif

__device__ __forceinline__ unsigned lds_byte_addr(const void* p) {
  // generic -> LDS(3) addrspacecast, then 32-bit LDS byte address
  return (unsigned)(unsigned long long)
         (__attribute__((address_space(3))) const char*)p;
}

// Issue one TDM DMA: copy nelem_f16 f16 elements (1 row) global -> LDS.
// D# built per CDNA5 ISA (groups 0/1; groups 2/3 zero).
__device__ __forceinline__ void tdm_load_1d(const void* gsrc, unsigned ldsDst,
                                            unsigned nelem_f16) {
#if USE_TDM
  unsigned long long ga = (unsigned long long)gsrc;
  u32x4 g0;
  g0[0] = 1u;                                   // count=1, user descriptor
  g0[1] = ldsDst;                               // lds_addr [63:32]
  g0[2] = (unsigned)(ga & 0xffffffffu);         // global_addr [95:64]
  g0[3] = (unsigned)((ga >> 32) & 0x01ffffffu)  // global_addr [120:96]
        | (2u << 30);                           // type=2 ("image") [127:126]
  i32x8 g1;
  g1[0] = (1 << 16);                            // data_size=1 (2 bytes/elem)
  g1[1] = (int)((nelem_f16 & 0xffffu) << 16);   // tensor_dim0 low16 @ [63:48]
  g1[2] = (int)((nelem_f16 >> 16) & 0xffffu)    // tensor_dim0 high16 @ [79:64]
        | (1 << 16);                            // tensor_dim1 = 1 @ [95:80]
  g1[3] = (int)((nelem_f16 & 0xffffu) << 16);   // tile_dim0 @ [127:112]
  g1[4] = 1;                                    // tile_dim1 = 1 @ [143:128]
  g1[5] = (int)nelem_f16;                       // tensor_dim0_stride [191:160]
  g1[6] = 0;
  g1[7] = 0;
  i32x4 z4 = {0, 0, 0, 0};
#  if TDM_6ARG
  i32x8 z8 = {0, 0, 0, 0, 0, 0, 0, 0};
  __builtin_amdgcn_tensor_load_to_lds(g0, g1, z4, z4, z8, 0);
#  else
  __builtin_amdgcn_tensor_load_to_lds(g0, g1, z4, z4, 0);
#  endif
  __builtin_amdgcn_s_wait_tensorcnt(0);
#endif
}

// ---------------- fragment loads (contiguous k order, A and B alike) ----------------
// A 16x32 f16: lanes 0-15 -> M=lane, K = 32*ks + {0..15}
//              lanes16-31 -> M=lane-16, K = 32*ks + {16..31}
__device__ __forceinline__ v16h load_a_frag(const _Float16* Arow, int ks, bool hi16) {
  int kb = ks * 32 + (hi16 ? 16 : 0);
  v8h lo = *(const v8h*)(Arow + kb);        // K = kb .. kb+7
  v8h hi = *(const v8h*)(Arow + kb + 8);    // K = kb+8 .. kb+15
  v16h a;
#pragma unroll
  for (int i = 0; i < 8; ++i) { a[i] = lo[i]; a[i + 8] = hi[i]; }
  return a;
}

// B 32x16 f16 (from BT stored [n][k] row-major in LDS): column n = lane%16,
// lanes 0-15 hold K = 32*ks+0..15, lanes 16-31 hold K = 32*ks+16..31.
__device__ __forceinline__ v16h load_b_frag(const _Float16* BT, int nrow, int ks,
                                            bool hi16) {
  int kb = ks * 32 + (hi16 ? 16 : 0);
  return *(const v16h*)(BT + (unsigned)nrow * D128 + kb);   // 32B contiguous
}

// ---------------- generic [M,128] x [128,128] WMMA GEMM --------------------
// C = A * BT^T (+bias) (opt relu); A f16 row-major lda=128, BT f16 [128][128].
template <bool HAS_BIAS, bool RELU, bool OUTF16>
__global__ void gemm_nk128(const _Float16* __restrict__ A,
                           const _Float16* __restrict__ BT,
                           const float* __restrict__ bias,
                           void* __restrict__ Cout, int M) {
  __shared__ _Float16 ldsBT[D128 * D128];     // 32KB weight tile, WG-shared

  // ---- stage B^T into LDS once per workgroup ------------------------------
#if USE_TDM
  if (threadIdx.x < 32) {                     // single TDM issue from wave 0
    tdm_load_1d(BT, lds_byte_addr(&ldsBT[0]), D128 * D128);
  }
#else
  for (int i = threadIdx.x; i < (D128 * D128) / 8; i += 256)
    ((v8h*)ldsBT)[i] = ((const v8h*)BT)[i];
#endif
  __syncthreads();                            // every wave participates

  const int lane = threadIdx.x & 31;
  const int wave = threadIdx.x >> 5;
  const int tile = blockIdx.x * (blockDim.x >> 5) + wave;
  const int m0 = tile * 16;
  if (m0 < M) {                               // wave-uniform: EXEC stays full
    int mrow = m0 + (lane & 15);
    if (mrow > M - 1) mrow = M - 1;           // clamp loads for ragged tail
    const _Float16* Arow = A + (size_t)mrow * D128;
    const bool hi16 = (lane & 16) != 0;
    const bool full = (m0 + 16 <= M);         // wave-uniform

    v16h a0 = load_a_frag(Arow, 0, hi16);
    v16h a1 = load_a_frag(Arow, 1, hi16);
    v16h a2 = load_a_frag(Arow, 2, hi16);
    v16h a3 = load_a_frag(Arow, 3, hi16);

    const int nlo = lane & 15;
    const int rbase = m0 + (hi16 ? 8 : 0);
#pragma unroll
    for (int nt = 0; nt < 8; ++nt) {
      const int ncol = nt * 16 + nlo;
      v8f acc = {};
      acc = __builtin_amdgcn_wmma_f32_16x16x32_f16(false, a0, false,
              load_b_frag(ldsBT, ncol, 0, hi16), (short)0, acc, false, false);
      acc = __builtin_amdgcn_wmma_f32_16x16x32_f16(false, a1, false,
              load_b_frag(ldsBT, ncol, 1, hi16), (short)0, acc, false, false);
      acc = __builtin_amdgcn_wmma_f32_16x16x32_f16(false, a2, false,
              load_b_frag(ldsBT, ncol, 2, hi16), (short)0, acc, false, false);
      acc = __builtin_amdgcn_wmma_f32_16x16x32_f16(false, a3, false,
              load_b_frag(ldsBT, ncol, 3, hi16), (short)0, acc, false, false);

      const float bb = HAS_BIAS ? bias[ncol] : 0.0f;   // compile-time select
      if (full) {                             // branch-free store path
#pragma unroll
        for (int r = 0; r < 8; ++r) {
          float v = acc[r] + bb;
          if (RELU) v = fmaxf(v, 0.0f);
          if (OUTF16) ((_Float16*)Cout)[(size_t)(rbase + r) * D128 + ncol] = (_Float16)v;
          else        ((float*)Cout)[(size_t)(rbase + r) * D128 + ncol] = v;
        }
      } else {
#pragma unroll
        for (int r = 0; r < 8; ++r) {
          const int row = rbase + r;
          if (row < M) {
            float v = acc[r] + bb;
            if (RELU) v = fmaxf(v, 0.0f);
            if (OUTF16) ((_Float16*)Cout)[(size_t)row * D128 + ncol] = (_Float16)v;
            else        ((float*)Cout)[(size_t)row * D128 + ncol] = v;
          }
        }
      }
    }
  }
}

// ---------------- weight prep: f32 -> f16, optional transpose --------------
// blockIdx.y selects which of the 6 weight matrices.
__global__ void prep_weights(const float* __restrict__ Wq, const float* __restrict__ Wk,
                             const float* __restrict__ Wv, const float* __restrict__ Wo,
                             const float* __restrict__ W1, const float* __restrict__ W2,
                             _Float16* __restrict__ Obase) {
  const int m = blockIdx.y;
  const float* src;
  switch (m) {
    case 0: src = Wq; break;
    case 1: src = Wk; break;
    case 2: src = Wv; break;
    case 3: src = Wo; break;
    case 4: src = W1; break;
    default: src = W2; break;
  }
  const bool transpose = (m != 1);            // Wk stays as-is (it IS B^T for U)
  const int i = blockIdx.x * 256 + threadIdx.x;
  if (i >= D128 * D128) return;
  const int r = i >> 7, c = i & 127;
  const float v = transpose ? src[c * D128 + r] : src[i];
  Obase[m * D128 * D128 + i] = (_Float16)v;
}

__global__ void convert_f32_f16(const float* __restrict__ X,
                                _Float16* __restrict__ O, long long n) {
  const long long i = (long long)blockIdx.x * 256 + threadIdx.x;
  if (i < n) O[i] = (_Float16)X[i];
}

// ---------------- attention core: one wave32 per node ----------------------
// scores[l,h] = scale * u_h . val_l ; online softmax ; wbar_h = sum a_l val_l;
// attn_out = concat_h(wbar_h Wv_h) + bv  (Wv via WvT[hk][d] f16).
__global__ void attn_kernel(const _Float16* __restrict__ Xh,
                            const _Float16* __restrict__ Uh,
                            const _Float16* __restrict__ WvT,
                            const float* __restrict__ bv,
                            const int* __restrict__ nbr,
                            const int* __restrict__ vlen,
                            _Float16* __restrict__ AttnOut, int N) {
  __shared__ float wbar[8][4][D128];          // 8 waves * 2KB = 16KB
  const int lane = threadIdx.x & 31;
  const int wave = threadIdx.x >> 5;
  const int n = blockIdx.x * 8 + wave;
  const bool live = (n < N);
  const int nc = live ? n : 0;

  __builtin_prefetch(nbr + (size_t)nc * 16, 0, 0);   // global_prefetch_b8

  const int d0 = lane * 4;                    // this lane owns cols d0..d0+3
  float uf[4];
  {
    v4h u4 = *(const v4h*)(Uh + (size_t)nc * D128 + d0);
#pragma unroll
    for (int j = 0; j < 4; ++j) uf[j] = (float)u4[j];
  }
  const int valid = vlen[nc];                 // mask: l <= valid (l=0 = self)

  float mh[4], sh[4], acc[4][4];
#pragma unroll
  for (int h = 0; h < 4; ++h) {
    mh[h] = -1e30f; sh[h] = 0.0f;
#pragma unroll
    for (int j = 0; j < 4; ++j) acc[h][j] = 0.0f;
  }
  const float scale = 0.17677669529663687f;   // 1/sqrt(KD=32)

  for (int l = 0; l <= 16; ++l) {
    const int src = (l == 0) ? nc : nbr[(size_t)nc * 16 + (l - 1)];
    v4h v4 = *(const v4h*)(Xh + (size_t)src * D128 + d0);
    float vv[4];
#pragma unroll
    for (int j = 0; j < 4; ++j) vv[j] = (float)v4[j];

    // per-lane partial dot; lane's 4 dims all belong to head (lane>>3)
    float part = uf[0]*vv[0] + uf[1]*vv[1] + uf[2]*vv[2] + uf[3]*vv[3];
    part += __shfl_xor(part, 1);
    part += __shfl_xor(part, 2);
    part += __shfl_xor(part, 4);              // 8-lane head-group reduction
    const bool masked = (l > valid);
#pragma unroll
    for (int h = 0; h < 4; ++h) {
      float s = __shfl(part, h * 8) * scale;  // broadcast head h's score
      if (masked) s = -1e9f;
      const float mn = fmaxf(mh[h], s);
      const float r = __expf(mh[h] - mn);
      const float p = __expf(s - mn);
      mh[h] = mn;
      sh[h] = sh[h] * r + p;
#pragma unroll
      for (int j = 0; j < 4; ++j) acc[h][j] = acc[h][j] * r + p * vv[j];
    }
  }

#pragma unroll
  for (int h = 0; h < 4; ++h) {
    const float inv = 1.0f / sh[h];
#pragma unroll
    for (int j = 0; j < 4; ++j) wbar[wave][h][d0 + j] = acc[h][j] * inv;
  }
  __syncthreads();                            // all waves reach (no early ret)

  // attn_out[col] = bv[col] + wbar[h(col)] . WvT[col][:]
#pragma unroll
  for (int j = 0; j < 4; ++j) {
    const int col = d0 + j;
    const int h = col >> 5;
    float s = bv[col];
    const _Float16* wr = WvT + (size_t)col * D128;
    const float* wb = wbar[wave][h];
#pragma unroll 2
    for (int d = 0; d < D128; d += 8) {
      v8h w8 = *(const v8h*)(wr + d);
#pragma unroll
      for (int t = 0; t < 8; ++t) s += wb[d + t] * (float)w8[t];
    }
    if (live) AttnOut[(size_t)n * D128 + col] = (_Float16)s;
  }
}

// ---------------------------------------------------------------------------
extern "C" void kernel_launch(void* const* d_in, const int* in_sizes, int n_in,
                              void* d_out, int out_size, void* d_ws, size_t ws_size,
                              hipStream_t stream) {
  const float* X    = (const float*)d_in[0];
  const int*   nbr  = (const int*)d_in[1];
  const int*   vlen = (const int*)d_in[2];
  const float* Wq   = (const float*)d_in[3];
  const float* bq   = (const float*)d_in[4];
  const float* Wk   = (const float*)d_in[5];
  /* bk (d_in[6]) cancels in softmax */
  const float* Wv   = (const float*)d_in[7];
  const float* bv   = (const float*)d_in[8];
  const float* Wo   = (const float*)d_in[9];
  const float* bo   = (const float*)d_in[10];
  const float* W1   = (const float*)d_in[11];
  const float* b1   = (const float*)d_in[12];
  const float* W2   = (const float*)d_in[13];
  const float* b2   = (const float*)d_in[14];
  float* out = (float*)d_out;
  const int N = in_sizes[0] / D128;

  char* ws = (char*)d_ws;
  const size_t nbuf = (size_t)N * D128 * sizeof(_Float16);   // 25.6 MB
  _Float16* Xh  = (_Float16*)(ws);             // X f16; later reused for h1
  _Float16* Qh  = (_Float16*)(ws + nbuf);      // Q f16; later reused for attn_out
  _Float16* Uh  = (_Float16*)(ws + 2 * nbuf);  // U f16; later reused for attn2
  _Float16* Wb  = (_Float16*)(ws + 3 * nbuf);  // 6 x 32KB f16 weights
  _Float16* WqT = Wb + 0 * 16384;
  _Float16* WkH = Wb + 1 * 16384;              // Wk as-is IS B^T for the U GEMM
  _Float16* WvT = Wb + 2 * 16384;
  _Float16* WoT = Wb + 3 * 16384;
  _Float16* W1T = Wb + 4 * 16384;
  _Float16* W2T = Wb + 5 * 16384;

  prep_weights<<<dim3(64, 6), 256, 0, stream>>>(Wq, Wk, Wv, Wo, W1, W2, Wb);

  const long long nx = (long long)N * D128;
  convert_f32_f16<<<(int)((nx + 255) / 256), 256, 0, stream>>>(X, Xh, nx);

  const int tiles  = (N + 15) / 16;
  const int blocks = (tiles + 7) / 8;          // 8 wave32 tiles per block

  // Q = Xh * Wq^T + bq  (f16)
  gemm_nk128<true , false, true ><<<blocks, 256, 0, stream>>>(Xh, WqT, bq, Qh, N);
  // U = Q * Wk          (f16)   [u_{n,d} = sum_hk q_{n,hk} Wk_{d,hk}]
  gemm_nk128<false, false, true ><<<blocks, 256, 0, stream>>>(Qh, WkH, nullptr, Uh, N);
  // attention: reads Xh,Uh -> writes attn_out into Qh (Q dead now)
  attn_kernel<<<(N + 7) / 8, 256, 0, stream>>>(Xh, Uh, WvT, bv, nbr, vlen, Qh, N);
  // attn2 = attn_out * Wo^T + bo -> Uh (U dead now)
  gemm_nk128<true , false, true ><<<blocks, 256, 0, stream>>>(Qh, WoT, bo, Uh, N);
  // h1 = relu(attn2 * W1^T + b1) -> Xh (Xh dead now)
  gemm_nk128<true , true , true ><<<blocks, 256, 0, stream>>>(Uh, W1T, b1, Xh, N);
  // out = h1 * W2^T + b2 (f32)
  gemm_nk128<true , false, false><<<blocks, 256, 0, stream>>>(Xh, W2T, b2, out, N);
}
